// EvolvedLoopLinear_21251498180720
// MI455X (gfx1250) — matrix operand order repaired and checked by reading, not compiled, not run on hardware
//
#include <hip/hip_runtime.h>
#include <hip/hip_bf16.h>
#include <stdint.h>

// ---------------------------------------------------------------------------
// out[M,N] = x[M,K] @ W[N,K]^T + 2*b[N],  M=N=K=4096, fp32 in/out.
// Pass 1: convert x, W to bf16 in d_ws (bandwidth-bound).
// Pass 2: bf16 WMMA GEMM, f32 accumulate. Block tile 256x128, K-tile 64.
//   - 8 waves, each owns a 64x64 tile = 4x4 WMMA accumulators (128 VGPRs);
//     8 fragment loads per 16 WMMAs -> 1:1 ds:wmma, 16 B/lane LDS per WMMA.
//   - CDNA5 async global->LDS staging (ASYNCcnt), TRIPLE buffered:
//     tile k+2 issued while computing tile k; s_wait_asynccnt 0xc ensures
//     tile k+1 resident (async loads complete in order; 12 ops/thread/tile).
//   - IOFFSET immediates (added to both LDS and global sides per ISA) keep
//     staging address math to one rolling pointer per matrix.
// Fallback (ws too small): convert-in-kernel GEMM (128x128, double buffer).
// ---------------------------------------------------------------------------

typedef __attribute__((ext_vector_type(16))) __bf16     bf16x16;
typedef __attribute__((ext_vector_type(8)))  float      floatx8;
typedef __attribute__((ext_vector_type(4)))  unsigned   uint4v;

#define M_DIM 4096
#define N_DIM 4096
#define K_DIM 4096
#define BTM   256            // block tile M (async kernel)
#define BTN   128            // block tile N (async kernel)
#define KT2   64             // K tile for async kernel
#define NT2   (K_DIM / KT2)
#define BT1   128            // block tile for fallback kernel
#define KT1   32             // K tile for fallback kernel
#define NT1   (K_DIM / KT1)

union FragU { uint4v u[2]; bf16x16 v; };

__device__ __forceinline__ unsigned short f2bf(float f) {
    unsigned u = __float_as_uint(f);
    u += 0x7FFFu + ((u >> 16) & 1u);   // round-to-nearest-even
    return (unsigned short)(u >> 16);
}
__device__ __forceinline__ unsigned pk2(float a, float b) {
    return (unsigned)f2bf(a) | ((unsigned)f2bf(b) << 16);
}
__device__ __forceinline__ unsigned lds_off32(const void* p) {
    return (unsigned)(uintptr_t)p;    // low 32 bits of LDS aperture = byte offset
}

// ---- CDNA5 async global->LDS copies; IOFFSET applies to LDS and global.
__device__ __forceinline__ void stage_async4(unsigned lds_off, const unsigned short* g) {
    asm volatile(
        "s_clause 0x3\n\t"
        "global_load_async_to_lds_b128 %0, %1, off\n\t"
        "global_load_async_to_lds_b128 %0, %1, off offset:16\n\t"
        "global_load_async_to_lds_b128 %0, %1, off offset:32\n\t"
        "global_load_async_to_lds_b128 %0, %1, off offset:48"
        :: "v"(lds_off), "v"(g) : "memory");
}
__device__ __forceinline__ void stage_async8(unsigned lds_off, const unsigned short* g) {
    asm volatile(
        "s_clause 0x7\n\t"
        "global_load_async_to_lds_b128 %0, %1, off\n\t"
        "global_load_async_to_lds_b128 %0, %1, off offset:16\n\t"
        "global_load_async_to_lds_b128 %0, %1, off offset:32\n\t"
        "global_load_async_to_lds_b128 %0, %1, off offset:48\n\t"
        "global_load_async_to_lds_b128 %0, %1, off offset:64\n\t"
        "global_load_async_to_lds_b128 %0, %1, off offset:80\n\t"
        "global_load_async_to_lds_b128 %0, %1, off offset:96\n\t"
        "global_load_async_to_lds_b128 %0, %1, off offset:112"
        :: "v"(lds_off), "v"(g) : "memory");
}
#define WAIT_ASYNC_0()  asm volatile("s_wait_asynccnt 0x0" ::: "memory")
#define WAIT_ASYNC_12() asm volatile("s_wait_asynccnt 0xc" ::: "memory")

// ---------------- Pass 1: f32 -> bf16 conversion ---------------------------
__global__ __launch_bounds__(256)
void cvt_f32_bf16_kernel(const float* __restrict__ src,
                         unsigned short* __restrict__ dst, int n) {
    size_t i = ((size_t)blockIdx.x * blockDim.x + threadIdx.x) * 8;
    if (i >= (size_t)n) return;
    float4 a = *(const float4*)(src + i);
    float4 c = *(const float4*)(src + i + 4);
    uint4v o = { pk2(a.x, a.y), pk2(a.z, a.w), pk2(c.x, c.y), pk2(c.z, c.w) };
    *(uint4v*)(dst + i) = o;
}

// ---------------- Pass 2: bf16 GEMM, async triple-buffered staging ---------
__global__ __launch_bounds__(256, 1)
void gemm_bf16_async_kernel(const unsigned short* __restrict__ xa,
                            const unsigned short* __restrict__ wb,
                            const float* __restrict__ bias,
                            float* __restrict__ out) {
    __shared__ unsigned short shA[3][BTM * KT2];   // 32 KB each
    __shared__ unsigned short shB[3][BTN * KT2];   // 16 KB each

    const int tid  = threadIdx.x;
    const int lane = tid & 31;
    const int wv   = tid >> 5;
    const int wm   = wv & 3;          // wave M position: wm*64
    const int wn   = wv >> 2;         // wave N position: wn*64
    const int r    = lane & 15;
    const int hi   = lane >> 4;
    const int kbA  = hi * 8;          // A-frag K base (ISA 16-bit A layout)
    const int kbB  = hi * 8;          // B-frag K base (same K order as A)

    const int tileM = blockIdx.y * BTM;
    const int tileN = blockIdx.x * BTN;

    // staging: A = one full 64-elem row per thread; B = 32-elem half row
    const unsigned short* gA = xa + (size_t)(tileM + tid) * K_DIM;
    const int sbaseA = tid * KT2;
    const int brow = tid >> 1;
    const int bcol = (tid & 1) * 32;
    const unsigned short* gB = wb + (size_t)(tileN + brow) * K_DIM + bcol;
    const int sbaseB = brow * KT2 + bcol;

    // rotating buffer state: staging LDS offsets + compute read pointers
    unsigned aoff_c = lds_off32(&shA[0][sbaseA]);
    unsigned aoff_n = lds_off32(&shA[1][sbaseA]);
    unsigned aoff_f = lds_off32(&shA[2][sbaseA]);
    unsigned boff_c = lds_off32(&shB[0][sbaseB]);
    unsigned boff_n = lds_off32(&shB[1][sbaseB]);
    unsigned boff_f = lds_off32(&shB[2][sbaseB]);
    const unsigned short *sa_c = shA[0], *sa_n = shA[1], *sa_f = shA[2];
    const unsigned short *sb_c = shB[0], *sb_n = shB[1], *sb_f = shB[2];

    // rolling global pointers for the issue stage (tile k+2)
    const unsigned short* gAf = gA + 2 * KT2;
    const unsigned short* gBf = gB + 2 * KT2;

    floatx8 acc[4][4] = {};   // 128 VGPRs of f32 accumulators

    // prologue: issue tiles 0 and 1; wait for tile 0 (<=12 outstanding)
    stage_async8(aoff_c, gA);
    stage_async4(boff_c, gB);
    stage_async8(aoff_n, gA + KT2);
    stage_async4(boff_n, gB + KT2);
    WAIT_ASYNC_12();
    __syncthreads();

    for (int kt = 0; kt < NT2; ++kt) {
        // issue tile kt+2 into the far buffer (freed by barrier at end of kt-1)
        if (kt + 2 < NT2) {
            stage_async8(aoff_f, gAf);
            stage_async4(boff_f, gBf);
            gAf += KT2;
            gBf += KT2;
            __builtin_prefetch(gAf, 0, 1);   // L2 prefetch one more tile ahead
            __builtin_prefetch(gBf, 0, 1);
        }

        // two 32-deep K slices per tile -> 32 WMMAs per barrier period
        #pragma unroll
        for (int kk = 0; kk < 2; ++kk) {
            const int ko = kk * 32;

            bf16x16 afrag[4];
            #pragma unroll
            for (int im = 0; im < 4; ++im) {
                const int rowA = wm * 64 + im * 16 + r;
                FragU f;
                f.u[0] = *(const uint4v*)&sa_c[rowA * KT2 + ko + kbA];
                f.u[1] = *(const uint4v*)&sa_c[rowA * KT2 + ko + kbA + 16];
                afrag[im] = f.v;
            }
            bf16x16 bfrag[4];
            #pragma unroll
            for (int in = 0; in < 4; ++in) {
                const int rowB = wn * 64 + in * 16 + r;
                FragU f;
                f.u[0] = *(const uint4v*)&sb_c[rowB * KT2 + ko + kbB];
                f.u[1] = *(const uint4v*)&sb_c[rowB * KT2 + ko + kbB + 16];
                bfrag[in] = f.v;
            }

            #pragma unroll
            for (int im = 0; im < 4; ++im)
                #pragma unroll
                for (int in = 0; in < 4; ++in)
                    acc[im][in] = __builtin_amdgcn_wmma_f32_16x16x32_bf16(
                        false, afrag[im], false, bfrag[in],
                        (short)0, acc[im][in], false, false);
        }

        // ensure tile kt+1 is resident, then hand buffers around
        if (kt + 2 < NT2) {
            WAIT_ASYNC_12();      // in-order completion: older tile's 12 ops done
            __syncthreads();
        } else if (kt + 1 < NT2) {
            WAIT_ASYNC_0();       // nothing new issued; drain fully
            __syncthreads();
        }

        // rotate cur <- next <- far <- cur
        unsigned tu;
        const unsigned short* tp;
        tu = aoff_c; aoff_c = aoff_n; aoff_n = aoff_f; aoff_f = tu;
        tu = boff_c; boff_c = boff_n; boff_n = boff_f; boff_f = tu;
        tp = sa_c; sa_c = sa_n; sa_n = sa_f; sa_f = tp;
        tp = sb_c; sb_c = sb_n; sb_n = sb_f; sb_f = tp;
    }

    // epilogue: out[m][n] = acc + 2*b[n]
    #pragma unroll
    for (int in = 0; in < 4; ++in) {
        const int n  = tileN + wn * 64 + in * 16 + r;
        const float bv = 2.0f * bias[n];
        #pragma unroll
        for (int im = 0; im < 4; ++im) {
            const int m0 = tileM + wm * 64 + im * 16 + hi * 8;
            #pragma unroll
            for (int v = 0; v < 8; ++v)
                out[(size_t)(m0 + v) * N_DIM + n] = acc[im][in][v] + bv;
        }
    }
}

// ---------------- Fallback: convert-in-kernel GEMM (round-1 style) ---------
__global__ __launch_bounds__(256, 2)
void gemm_f32src_fallback_kernel(const float* __restrict__ x,
                                 const float* __restrict__ W,
                                 const float* __restrict__ b,
                                 float* __restrict__ out) {
    __shared__ unsigned short shA[2][BT1 * KT1];
    __shared__ unsigned short shB[2][BT1 * KT1];

    const int tid  = threadIdx.x;
    const int lane = tid & 31;
    const int wv   = tid >> 5;
    const int wm   = wv & 3;
    const int wn   = wv >> 2;
    const int r    = lane & 15;
    const int hi   = lane >> 4;
    const int kbA  = hi * 8;
    const int kbB  = hi * 8;

    const int tileM = blockIdx.y * BT1;
    const int tileN = blockIdx.x * BT1;

    const int srow = tid >> 1;
    const int scol = (tid & 1) * 16;
    const float* gA = x + (size_t)(tileM + srow) * K_DIM + scol;
    const float* gB = W + (size_t)(tileN + srow) * K_DIM + scol;
    const int sbase = srow * KT1 + scol;

    floatx8 acc[2][4] = {};
    float4 ra[4], rb[4];

    {
        const float4* pa = (const float4*)gA;
        const float4* pb = (const float4*)gB;
        #pragma unroll
        for (int i = 0; i < 4; ++i) { ra[i] = pa[i]; rb[i] = pb[i]; }
        uint4v a0 = { pk2(ra[0].x, ra[0].y), pk2(ra[0].z, ra[0].w),
                      pk2(ra[1].x, ra[1].y), pk2(ra[1].z, ra[1].w) };
        uint4v a1 = { pk2(ra[2].x, ra[2].y), pk2(ra[2].z, ra[2].w),
                      pk2(ra[3].x, ra[3].y), pk2(ra[3].z, ra[3].w) };
        uint4v b0 = { pk2(rb[0].x, rb[0].y), pk2(rb[0].z, rb[0].w),
                      pk2(rb[1].x, rb[1].y), pk2(rb[1].z, rb[1].w) };
        uint4v b1 = { pk2(rb[2].x, rb[2].y), pk2(rb[2].z, rb[2].w),
                      pk2(rb[3].x, rb[3].y), pk2(rb[3].z, rb[3].w) };
        *(uint4v*)&shA[0][sbase]     = a0;
        *(uint4v*)&shA[0][sbase + 8] = a1;
        *(uint4v*)&shB[0][sbase]     = b0;
        *(uint4v*)&shB[0][sbase + 8] = b1;
    }
    __syncthreads();

    for (int kt = 0; kt < NT1; ++kt) {
        const int buf = kt & 1;
        if (kt + 1 < NT1) {
            const float4* pa = (const float4*)(gA + (size_t)(kt + 1) * KT1);
            const float4* pb = (const float4*)(gB + (size_t)(kt + 1) * KT1);
            #pragma unroll
            for (int i = 0; i < 4; ++i) { ra[i] = pa[i]; rb[i] = pb[i]; }
        }

        const unsigned short* sa = shA[buf];
        const unsigned short* sb = shB[buf];
        bf16x16 afrag[2];
        #pragma unroll
        for (int im = 0; im < 2; ++im) {
            const int rowA = wm * 32 + im * 16 + r;
            FragU f;
            f.u[0] = *(const uint4v*)&sa[rowA * KT1 + kbA];
            f.u[1] = *(const uint4v*)&sa[rowA * KT1 + kbA + 16];
            afrag[im] = f.v;
        }
        bf16x16 bfrag[4];
        #pragma unroll
        for (int in = 0; in < 4; ++in) {
            const int rowB = wn * 64 + in * 16 + r;
            FragU f;
            f.u[0] = *(const uint4v*)&sb[rowB * KT1 + kbB];
            f.u[1] = *(const uint4v*)&sb[rowB * KT1 + kbB + 16];
            bfrag[in] = f.v;
        }
        #pragma unroll
        for (int im = 0; im < 2; ++im)
            #pragma unroll
            for (int in = 0; in < 4; ++in)
                acc[im][in] = __builtin_amdgcn_wmma_f32_16x16x32_bf16(
                    false, afrag[im], false, bfrag[in],
                    (short)0, acc[im][in], false, false);

        if (kt + 1 < NT1) {
            __syncthreads();
            const int nb = (kt + 1) & 1;
            uint4v a0 = { pk2(ra[0].x, ra[0].y), pk2(ra[0].z, ra[0].w),
                          pk2(ra[1].x, ra[1].y), pk2(ra[1].z, ra[1].w) };
            uint4v a1 = { pk2(ra[2].x, ra[2].y), pk2(ra[2].z, ra[2].w),
                          pk2(ra[3].x, ra[3].y), pk2(ra[3].z, ra[3].w) };
            uint4v b0 = { pk2(rb[0].x, rb[0].y), pk2(rb[0].z, rb[0].w),
                          pk2(rb[1].x, rb[1].y), pk2(rb[1].z, rb[1].w) };
            uint4v b1 = { pk2(rb[2].x, rb[2].y), pk2(rb[2].z, rb[2].w),
                          pk2(rb[3].x, rb[3].y), pk2(rb[3].z, rb[3].w) };
            *(uint4v*)&shA[nb][sbase]     = a0;
            *(uint4v*)&shA[nb][sbase + 8] = a1;
            *(uint4v*)&shB[nb][sbase]     = b0;
            *(uint4v*)&shB[nb][sbase + 8] = b1;
            __syncthreads();
        }
    }

    #pragma unroll
    for (int in = 0; in < 4; ++in) {
        const int n  = tileN + wn * 64 + in * 16 + r;
        const float bv = 2.0f * b[n];
        #pragma unroll
        for (int im = 0; im < 2; ++im) {
            const int m0 = tileM + wm * 32 + im * 16 + hi * 8;
            #pragma unroll
            for (int v = 0; v < 8; ++v)
                out[(size_t)(m0 + v) * N_DIM + n] = acc[im][in][v] + bv;
        }
    }
}

// ---------------------------------------------------------------------------
extern "C" void kernel_launch(void* const* d_in, const int* in_sizes, int n_in,
                              void* d_out, int out_size, void* d_ws, size_t ws_size,
                              hipStream_t stream) {
    const float* x = (const float*)d_in[0];
    const float* W = (const float*)d_in[1];
    const float* b = (const float*)d_in[2];
    float* out = (float*)d_out;

    const size_t needA = (size_t)M_DIM * K_DIM * sizeof(unsigned short);
    const size_t needB = (size_t)N_DIM * K_DIM * sizeof(unsigned short);

    if (ws_size >= needA + needB) {
        unsigned short* xa = (unsigned short*)d_ws;
        unsigned short* wb = xa + (size_t)M_DIM * K_DIM;
        const int nA = M_DIM * K_DIM;
        const int nB = N_DIM * K_DIM;
        cvt_f32_bf16_kernel<<<nA / 8 / 256, 256, 0, stream>>>(x, xa, nA);
        cvt_f32_bf16_kernel<<<nB / 8 / 256, 256, 0, stream>>>(W, wb, nB);
        dim3 grid(N_DIM / BTN, M_DIM / BTM);   // 32 x 16
        gemm_bf16_async_kernel<<<grid, dim3(256), 0, stream>>>(xa, wb, b, out);
    } else {
        dim3 grid(N_DIM / BT1, M_DIM / BT1);   // 32 x 32
        gemm_f32src_fallback_kernel<<<grid, dim3(256), 0, stream>>>(x, W, b, out);
    }
}
